// AttnDecoderRNN_58179626991671
// MI455X (gfx1250) — matrix operand order repaired and checked by reading, not compiled, not run on hardware
//
#include <hip/hip_runtime.h>
#include <hip/hip_bf16.h>
#include <math.h>

#define H 1024
#define V 32000
#define L 128
#define B 256

typedef __bf16 v16bf __attribute__((ext_vector_type(16)));
typedef __bf16 v8bf  __attribute__((ext_vector_type(8)));
typedef float  v8f   __attribute__((ext_vector_type(8)));

// Types for the async-LDS builtin: param is int __attribute__((vector_size(16)))*
// in AS1 (global) / AS3 (LDS), per the hipcc diagnostic.
typedef int v4i __attribute__((vector_size(16)));
typedef v4i __attribute__((address_space(1))) as1_v4i;
typedef v4i __attribute__((address_space(3))) as3_v4i;

static __device__ __forceinline__ __bf16 tobf(float x) { return (__bf16)x; }

// CDNA5 async global->LDS DMA (ASYNCcnt-tracked), if the toolchain exposes it.
#if defined(__has_builtin)
#  if __has_builtin(__builtin_amdgcn_global_load_async_to_lds_b128) && \
      __has_builtin(__builtin_amdgcn_s_wait_asynccnt)
#    define USE_ASYNC_LDS 1
#  endif
#endif
#ifndef USE_ASYNC_LDS
#  define USE_ASYNC_LDS 0
#endif

// ---------------------------------------------------------------------------
// C(256,N) = A(256,K) * W(N,K)^T + bias, optional relu.
// Block = 16 waves covering all of M=256; grid.x tiles N in 64s, so W streams
// from HBM exactly once. W tile is staged to LDS (async DMA double-buffer when
// available), converted to bf16, and fed to v_wmma_f32_16x16x32_bf16.
// ---------------------------------------------------------------------------
template <int RELU>
__global__ __launch_bounds__(512) void gemm256_wmma(
    const float* __restrict__ A, const float* __restrict__ W,
    const float* __restrict__ bias, float* __restrict__ C, int N, int K) {
#if USE_ASYNC_LDS
  // f32 ping-pong landing buffers for the async DMA (row stride 144B, 16B-aligned)
  __shared__ __align__(16) float Wf[2][64][36];
#endif
  // bf16 compute tile; stride 40 elems = 80B: 16B-aligned rows, 20-bank row
  // stride -> 16 consecutive rows hit 16 distinct banks for fragment reads.
  __shared__ __align__(16) __bf16 Bs[64][40];

  const int tid   = threadIdx.x;
  const int lane  = tid & 31;
  const int wave  = tid >> 5;        // 0..15
  const int m0    = wave << 4;       // wave's 16 output rows
  const int n0    = blockIdx.x << 6; // block's 64 output cols
  const int r     = lane & 15;
  const int hi    = lane >> 4;       // 0 or 1 (lane half)
  const int koffA = hi << 3;         // dense-A: K {0-7,16-23} / {8-15,24-31}
  const int koffB = hi << 3;         // dense-B: K {0-7,16-23} / {8-15,24-31}

  v8f acc[4] = {};                   // 4 C fragments: 16(M) x 64(N)

  const int wrow = tid >> 3;         // 0..63  (cooperative W stage)
  const int wc4  = (tid & 7) << 2;   // 0..28 step 4

#if USE_ASYNC_LDS
  int ping = 0;
  {  // preamble: DMA tile k0=0 into Wf[0] (one b128 per thread, no VGPR data)
    float* src = const_cast<float*>(W) + (size_t)(n0 + wrow) * K + wc4;
    __builtin_amdgcn_global_load_async_to_lds_b128(
        (as1_v4i*)src, (as3_v4i*)&Wf[0][wrow][wc4], 0, 0);
  }
#endif

  for (int k0 = 0; k0 < K; k0 += 32) {
#if USE_ASYNC_LDS
    __builtin_amdgcn_s_wait_asynccnt(0);  // tile k0 landed (this wave's part)
    __syncthreads();                      // all parts landed; prev wmma done with Bs
    {  // convert my 4 staged floats f32 -> bf16 compute tile
      const float4 v = *(const float4*)&Wf[ping][wrow][wc4];
      __bf16* d = &Bs[wrow][wc4];
      d[0] = tobf(v.x); d[1] = tobf(v.y); d[2] = tobf(v.z); d[3] = tobf(v.w);
    }
    if (k0 + 32 < K) {  // kick DMA for tile k0+32 into the other buffer
      float* src = const_cast<float*>(W) + (size_t)(n0 + wrow) * K + (k0 + 32) + wc4;
      __builtin_amdgcn_global_load_async_to_lds_b128(
          (as1_v4i*)src, (as3_v4i*)&Wf[ping ^ 1][wrow][wc4], 0, 0);
    }
    ping ^= 1;
    __syncthreads();                      // Bs ready for compute
#else
    {  // synchronous stage: W[n0+row][k0..k0+32) f32 -> bf16 LDS
      const float* src = W + (size_t)(n0 + wrow) * K + k0 + wc4;
      const float4 v = *(const float4*)src;
      __bf16* d = &Bs[wrow][wc4];
      d[0] = tobf(v.x); d[1] = tobf(v.y); d[2] = tobf(v.z); d[3] = tobf(v.w);
      if (k0 + 32 < K) __builtin_prefetch(src + 32, 0, 1);  // global_prefetch_b8
    }
    __syncthreads();
#endif

    // ---- A fragment (16x32 bf16), dense 16-bit A layout per ISA table ----
    v16bf af;
    {
      const float* a = A + (size_t)(m0 + r) * K + k0 + koffA;
      const float4 p0 = *(const float4*)(a);
      const float4 p1 = *(const float4*)(a + 4);
      const float4 p2 = *(const float4*)(a + 16);
      const float4 p3 = *(const float4*)(a + 20);
      af[0] = tobf(p0.x); af[1] = tobf(p0.y); af[2]  = tobf(p0.z); af[3]  = tobf(p0.w);
      af[4] = tobf(p1.x); af[5] = tobf(p1.y); af[6]  = tobf(p1.z); af[7]  = tobf(p1.w);
      af[8] = tobf(p2.x); af[9] = tobf(p2.y); af[10] = tobf(p2.z); af[11] = tobf(p2.w);
      af[12] = tobf(p3.x); af[13] = tobf(p3.y); af[14] = tobf(p3.z); af[15] = tobf(p3.w);
    }

    // ---- preload all 4 shared B fragments, then WMMA back-to-back ----
    v8bf blo[4], bhi[4];
#pragma unroll
    for (int j = 0; j < 4; ++j) {
      const __bf16* bp = &Bs[(j << 4) + r][koffB];
      blo[j] = *(const v8bf*)(bp);        // ds_load_b128
      bhi[j] = *(const v8bf*)(bp + 16);   // ds_load_b128
    }
#pragma unroll
    for (int j = 0; j < 4; ++j) {
      v16bf bfrag;
#pragma unroll
      for (int i = 0; i < 8; ++i) { bfrag[i] = blo[j][i]; bfrag[8 + i] = bhi[j][i]; }
      acc[j] = __builtin_amdgcn_wmma_f32_16x16x32_bf16(
          false, af, false, bfrag, (short)0, acc[j], false, false);
    }
#if !USE_ASYNC_LDS
    __syncthreads();
#endif
  }

  // ---- epilogue: C VGPR layout lane&15=N, (lane>=16?8:0)+i = M ----
  const int mbase = m0 + (hi << 3);
#pragma unroll
  for (int j = 0; j < 4; ++j) {
    const int n = n0 + (j << 4) + r;
    const float bv = bias[n];
#pragma unroll
    for (int i = 0; i < 8; ++i) {
      float v = acc[j][i] + bv;
      if (RELU) v = fmaxf(v, 0.f);
      C[(size_t)(mbase + i) * N + n] = v;
    }
  }
}

// ---------------------------------------------------------------------------
// Embedding gather + concat buffers: cat0 = [emb | h0], cat1[:, :H] = emb.
// ---------------------------------------------------------------------------
__global__ __launch_bounds__(256) void prep(
    const int* __restrict__ ids, const float* __restrict__ hidden,
    const float* __restrict__ emb, float* __restrict__ cat0,
    float* __restrict__ cat1) {
  const int b = blockIdx.x;
  const size_t erow = (size_t)ids[b] * H;
  const size_t c = (size_t)b * 2 * H;
  for (int i = threadIdx.x; i < H; i += 256) {
    const float e = emb[erow + i];
    cat0[c + i] = e;
    cat1[c + i] = e;
    cat0[c + H + i] = hidden[(size_t)b * H + i];
  }
}

// ---------------------------------------------------------------------------
// In-place softmax over L=128 per row (one block / row).
// ---------------------------------------------------------------------------
__global__ __launch_bounds__(128) void softmax_rows_L(float* __restrict__ aw) {
  __shared__ float red[L];
  const int b = blockIdx.x, t = threadIdx.x;
  const float v = aw[(size_t)b * L + t];
  red[t] = v;
  __syncthreads();
  for (int s = 64; s > 0; s >>= 1) {
    if (t < s) red[t] = fmaxf(red[t], red[t + s]);
    __syncthreads();
  }
  const float m = red[0];
  __syncthreads();
  const float e = __expf(v - m);
  red[t] = e;
  __syncthreads();
  for (int s = 64; s > 0; s >>= 1) {
    if (t < s) red[t] += red[t + s];
    __syncthreads();
  }
  aw[(size_t)b * L + t] = e / red[0];
}

// ---------------------------------------------------------------------------
// attn_applied[b,h] = sum_l w[b,l] * enc[b,l,h]  -> cat1[:, H:2H].
// Streams the 134 MB encoder tensor once; non-temporal so it doesn't evict
// the L2-resident activations/weights.
// ---------------------------------------------------------------------------
__global__ __launch_bounds__(256) void attn_apply(
    const float* __restrict__ aw, const float* __restrict__ enc,
    float* __restrict__ cat1) {
  __shared__ float w[L];
  const int b = blockIdx.y, t = threadIdx.x;
  if (t < L) w[t] = aw[(size_t)b * L + t];
  __syncthreads();
  const int h = (blockIdx.x << 8) + t;
  const float* e = enc + (size_t)b * L * H + h;
  float acc = 0.f;
#pragma unroll 8
  for (int l = 0; l < L; ++l)
    acc = fmaf(w[l], __builtin_nontemporal_load(e + (size_t)l * H), acc);
  cat1[(size_t)b * 2 * H + H + h] = acc;
}

// ---------------------------------------------------------------------------
// GRU gate math (PyTorch r,z,n layout). Biases already folded in via GEMM.
// ---------------------------------------------------------------------------
__global__ __launch_bounds__(256) void gru_step(
    const float* __restrict__ gx, const float* __restrict__ gh,
    const float* __restrict__ h0, float* __restrict__ hnew) {
  const int idx = blockIdx.x * 256 + threadIdx.x;
  const int b = idx >> 10, j = idx & (H - 1);
  const float* gxr = gx + (size_t)b * 3 * H;
  const float* ghr = gh + (size_t)b * 3 * H;
  const float r = 1.f / (1.f + __expf(-(gxr[j] + ghr[j])));
  const float z = 1.f / (1.f + __expf(-(gxr[j + H] + ghr[j + H])));
  const float n = tanhf(gxr[j + 2 * H] + r * ghr[j + 2 * H]);
  const float h = h0[idx];
  hnew[idx] = (1.f - z) * n + z * h;
}

// ---------------------------------------------------------------------------
// In-place log_softmax over V=32000 per row (one block / row).
// ---------------------------------------------------------------------------
__global__ __launch_bounds__(256) void log_softmax_rows(float* __restrict__ lg) {
  __shared__ float red[256];
  const int b = blockIdx.x, t = threadIdx.x;
  float* row = lg + (size_t)b * V;
  float m = -3.4e38f;
  for (int i = t; i < V; i += 256) m = fmaxf(m, row[i]);
  red[t] = m;
  __syncthreads();
  for (int s = 128; s > 0; s >>= 1) {
    if (t < s) red[t] = fmaxf(red[t], red[t + s]);
    __syncthreads();
  }
  m = red[0];
  __syncthreads();
  float sum = 0.f;
  for (int i = t; i < V; i += 256) sum += __expf(row[i] - m);
  red[t] = sum;
  __syncthreads();
  for (int s = 128; s > 0; s >>= 1) {
    if (t < s) red[t] += red[t + s];
    __syncthreads();
  }
  const float lse = m + __logf(red[0]);
  for (int i = t; i < V; i += 256) row[i] -= lse;
}

// ---------------------------------------------------------------------------
extern "C" void kernel_launch(void* const* d_in, const int* in_sizes, int n_in,
                              void* d_out, int out_size, void* d_ws,
                              size_t ws_size, hipStream_t stream) {
  (void)in_sizes; (void)n_in; (void)out_size; (void)ws_size;
  const int*   ids    = (const int*)d_in[0];
  const float* hidden = (const float*)d_in[1];   // (1,B,H)
  const float* enc    = (const float*)d_in[2];   // (B,L,H)
  const float* embt   = (const float*)d_in[3];   // (V,H)
  const float* attn_w = (const float*)d_in[4];   // (L,2H)
  const float* attn_b = (const float*)d_in[5];   // (L)
  const float* comb_w = (const float*)d_in[6];   // (H,2H)
  const float* comb_b = (const float*)d_in[7];   // (H)
  const float* w_ih   = (const float*)d_in[8];   // (3H,H)
  const float* b_ih   = (const float*)d_in[9];
  const float* w_hh   = (const float*)d_in[10];  // (3H,H)
  const float* b_hh   = (const float*)d_in[11];
  const float* out_w  = (const float*)d_in[12];  // (V,H)
  const float* out_b  = (const float*)d_in[13];  // (V)

  float* logp = (float*)d_out;              // B*V   (raw logits, then in-place logsoftmax)
  float* hnew = logp + (size_t)B * V;       // B*H
  float* aw   = hnew + (size_t)B * H;       // B*L   (raw attn logits, then in-place softmax)

  float* ws   = (float*)d_ws;
  float* cat0 = ws;                         // B*2H  [emb | h0]
  float* cat1 = cat0 + (size_t)B * 2 * H;   // B*2H  [emb | attn_applied]
  float* xbuf = cat1 + (size_t)B * 2 * H;   // B*H   combine output
  float* gx   = xbuf + (size_t)B * H;       // B*3H
  float* gh   = gx + (size_t)B * 3 * H;     // B*3H

  prep<<<B, 256, 0, stream>>>(ids, hidden, embt, cat0, cat1);
  gemm256_wmma<0><<<L / 64, 512, 0, stream>>>(cat0, attn_w, attn_b, aw, L, 2 * H);
  softmax_rows_L<<<B, L, 0, stream>>>(aw);
  attn_apply<<<dim3(H / 256, B), 256, 0, stream>>>(aw, enc, cat1);
  gemm256_wmma<1><<<H / 64, 512, 0, stream>>>(cat1, comb_w, comb_b, xbuf, H, 2 * H);
  gemm256_wmma<0><<<3 * H / 64, 512, 0, stream>>>(xbuf, w_ih, b_ih, gx, 3 * H, H);
  gemm256_wmma<0><<<3 * H / 64, 512, 0, stream>>>(hidden, w_hh, b_hh, gh, 3 * H, H);
  gru_step<<<(B * H) / 256, 256, 0, stream>>>(gx, gh, hidden, hnew);
  gemm256_wmma<0><<<V / 64, 512, 0, stream>>>(hnew, out_w, out_b, logp, V, H);
  log_softmax_rows<<<B, 256, 0, stream>>>(logp);
}
